// GraphAttentionLayerV2_89111981457766
// MI455X (gfx1250) — compile-verified, matrix operand order checked by reading, not run on hardware
//
#include <hip/hip_runtime.h>
#include <hip/hip_bf16.h>
#include <math.h>

typedef __attribute__((ext_vector_type(16))) __bf16 v16bf;
typedef __attribute__((ext_vector_type(4)))  __bf16 v4bf;
typedef __attribute__((ext_vector_type(8)))  float  v8f;
typedef __attribute__((ext_vector_type(4)))  float  v4f;

#define BATCH 32
#define NN 1024
#define DD 128
#define LEAKY 0.2f
#define LN_EPS 1e-14f

// ---------------------------------------------------------------------------
// Kernel 1: u = x@a_i, v = x@a_j per (b,i); separable LN stats:
//   mean(e) = mu_u + mu_v ; var(e) = var_u + var_v  (cross term == 0)
// Writes P[b,i] = u*rstd ; Q[b,j] = (v - mean)*rstd
// ---------------------------------------------------------------------------
__global__ __launch_bounds__(256) void k_stats(const float* __restrict__ x,
                                               const float* __restrict__ a,
                                               float* __restrict__ P,
                                               float* __restrict__ Q) {
  const int b = blockIdx.x;
  const int tid = threadIdx.x, lane = tid & 31, wave = tid >> 5;
  __shared__ float us[NN], vs[NN];
  __shared__ float red[32];
  __shared__ float stat[2];

  float a0[4], a1[4];
#pragma unroll
  for (int q = 0; q < 4; ++q) {
    a0[q] = a[lane + 32 * q];
    a1[q] = a[DD + lane + 32 * q];
  }
  for (int i = wave; i < NN; i += 8) {
    const float* xp = x + ((size_t)b * NN + i) * DD;
    float su = 0.f, sv = 0.f;
#pragma unroll
    for (int q = 0; q < 4; ++q) {
      float xv = xp[lane + 32 * q];
      su = fmaf(xv, a0[q], su);
      sv = fmaf(xv, a1[q], sv);
    }
#pragma unroll
    for (int off = 16; off >= 1; off >>= 1) {
      su += __shfl_xor(su, off, 32);
      sv += __shfl_xor(sv, off, 32);
    }
    if (lane == 0) { us[i] = su; vs[i] = sv; }
  }
  __syncthreads();

  float s1 = 0.f, s2 = 0.f, s3 = 0.f, s4 = 0.f;
  for (int i = tid; i < NN; i += 256) {
    float u = us[i], v = vs[i];
    s1 += u; s2 += u * u; s3 += v; s4 += v * v;
  }
#pragma unroll
  for (int off = 16; off >= 1; off >>= 1) {
    s1 += __shfl_xor(s1, off, 32);
    s2 += __shfl_xor(s2, off, 32);
    s3 += __shfl_xor(s3, off, 32);
    s4 += __shfl_xor(s4, off, 32);
  }
  if (lane == 0) {
    red[wave * 4 + 0] = s1; red[wave * 4 + 1] = s2;
    red[wave * 4 + 2] = s3; red[wave * 4 + 3] = s4;
  }
  __syncthreads();
  if (tid == 0) {
    float t1 = 0.f, t2 = 0.f, t3 = 0.f, t4 = 0.f;
#pragma unroll
    for (int w = 0; w < 8; ++w) {
      t1 += red[w * 4 + 0]; t2 += red[w * 4 + 1];
      t3 += red[w * 4 + 2]; t4 += red[w * 4 + 3];
    }
    float mu_u = t1 * (1.f / NN), mu_v = t3 * (1.f / NN);
    float var = (t2 * (1.f / NN) - mu_u * mu_u) + (t4 * (1.f / NN) - mu_v * mu_v);
    stat[0] = mu_u + mu_v;
    stat[1] = rsqrtf(var + LN_EPS);
  }
  __syncthreads();
  const float mean = stat[0], rstd = stat[1];
  for (int i = tid; i < NN; i += 256) {
    P[b * NN + i] = us[i] * rstd;
    Q[b * NN + i] = (vs[i] - mean) * rstd;
  }
}

// ---------------------------------------------------------------------------
// Kernel 2: per (b, j): online max/sum over i of exp(lrelu((P_i+Q_j)*g+bt))
// One thread per 4 consecutive columns -> b128 gamma/beta loads, L2-resident.
// ---------------------------------------------------------------------------
__global__ __launch_bounds__(256) void k_cols(const float* __restrict__ gamma,
                                              const float* __restrict__ beta,
                                              const float* __restrict__ P,
                                              const float* __restrict__ Q,
                                              float* __restrict__ CM,
                                              float* __restrict__ CI) {
  const int b = blockIdx.x;
  const int j0 = threadIdx.x * 4;
  __shared__ float Ps[NN];
  for (int i = threadIdx.x; i < NN; i += 256) Ps[i] = P[b * NN + i];
  __syncthreads();

  v4f qj = *(const v4f*)&Q[b * NN + j0];
  float m[4] = {-INFINITY, -INFINITY, -INFINITY, -INFINITY};
  float s[4] = {0.f, 0.f, 0.f, 0.f};
  for (int i = 0; i < NN; ++i) {
    v4f g  = *(const v4f*)&gamma[(size_t)i * NN + j0];
    v4f bt = *(const v4f*)&beta [(size_t)i * NN + j0];
    float pi = Ps[i];
#pragma unroll
    for (int c = 0; c < 4; ++c) {
      float t = fmaf(pi + qj[c], g[c], bt[c]);
      t = (t >= 0.f) ? t : LEAKY * t;
      float nm = fmaxf(m[c], t);
      s[c] = s[c] * __expf(m[c] - nm) + __expf(t - nm);
      m[c] = nm;
    }
  }
  v4f om, oi;
#pragma unroll
  for (int c = 0; c < 4; ++c) { om[c] = m[c]; oi[c] = 1.f / s[c]; }
  *(v4f*)&CM[b * NN + j0] = om;
  *(v4f*)&CI[b * NN + j0] = oi;
}

// ---------------------------------------------------------------------------
// Kernel 3: h[b,i,:] = sigmoid( sum_j exp(e~ - cmax_j) * (x[b,j,:]/csum_j) )
// Block = 8 waves, owns 128 i-rows x 128 d for one batch; K loop of 32.
// A (alpha) and B (scaled x) tiles are built directly in the WMMA per-lane
// bf16 operand layout in LDS; each wave issues 8x v_wmma_f32_16x16x32_bf16.
// Producer uses b128 global loads; A-tile stores are packed ds_store_b64.
// ---------------------------------------------------------------------------
__global__ __launch_bounds__(256) void k_gemm(const float* __restrict__ x,
                                              const float* __restrict__ gamma,
                                              const float* __restrict__ beta,
                                              const float* __restrict__ P,
                                              const float* __restrict__ Q,
                                              const float* __restrict__ CM,
                                              const float* __restrict__ CI,
                                              float* __restrict__ out) {
  const int b  = blockIdx.y;
  const int I0 = blockIdx.x * 128;
  const int tid = threadIdx.x, lane = tid & 31, wave = tid >> 5;

  __shared__ __align__(32) __bf16 Wop[8 * 32 * 16];  // A operands, 8 KB
  __shared__ __align__(32) __bf16 Yop[8 * 32 * 16];  // B operands, 8 KB
  __shared__ float Ps[128];
  __shared__ float qs[32], ms[32], ivs[32];

  if (tid < 128) Ps[tid] = P[b * NN + I0 + tid];

  v8f acc[8] = {};

  for (int kk = 0; kk < 32; ++kk) {
    const int j0 = kk * 32;
    if (tid < 32) {  // no hazard vs. previous consumer (touches only qs/ms/ivs)
      qs[tid]  = Q [b * NN + j0 + tid];
      ms[tid]  = CM[b * NN + j0 + tid];
      ivs[tid] = CI[b * NN + j0 + tid];
    }
    __syncthreads();  // qs/ms/ivs visible; prev consumer done with Wop/Yop

    // ---- producer: A tile = exp weights, 128 x 32, b128 loads, b64 stores --
#pragma unroll
    for (int r = 0; r < 4; ++r) {
      int idx = (r * 256 + tid) * 4;        // 4 consecutive jl per thread
      int il = idx >> 5, jl = idx & 31;
      const float* gp = &gamma[(size_t)(I0 + il) * NN + (j0 + jl)];
      const float* bp = &beta [(size_t)(I0 + il) * NN + (j0 + jl)];
      v4f g  = *(const v4f*)gp;
      v4f bt = *(const v4f*)bp;
      float pi = Ps[il];
      v4bf w4;
#pragma unroll
      for (int c = 0; c < 4; ++c) {
        float t = fmaf(pi + qs[jl + c], g[c], bt[c]);
        t = (t >= 0.f) ? t : LEAKY * t;
        w4[c] = (__bf16)__expf(t - ms[jl + c]);
      }
      int h  = (jl >> 3) & 1;               // constant across the 4-run
      int ln = (il & 15) + (h << 4);
      int e  = ((jl >> 4) << 3) | (jl & 7); // e..e+3 contiguous
      *(v4bf*)&Wop[(((il >> 4) << 5) + ln) * 16 + e] = w4;
    }
    // ---- producer: B tile = x[b, j0:j0+32, :]/csum, b128 loads ------------
#pragma unroll
    for (int r = 0; r < 4; ++r) {
      int idx = (r * 256 + tid) * 4;        // 4 consecutive d per thread
      int jl = idx >> 7, d0 = idx & 127;
      const float* xp = &x[((size_t)b * NN + j0 + jl) * DD + d0];
      if (kk < 31) __builtin_prefetch(xp + 32 * DD, 0, 1);  // next K chunk
      v4f xv = *(const v4f*)xp;
      float iv = ivs[jl];
      int h = (jl >> 3) & 1;
      int e = ((jl >> 4) << 3) | (jl & 7);
      int tle = (d0 >> 4) << 5;             // constant across the 4-run
#pragma unroll
      for (int c = 0; c < 4; ++c) {
        int ln = ((d0 + c) & 15) + (h << 4);
        Yop[(tle + ln) * 16 + e] = (__bf16)(xv[c] * iv);
      }
    }
    __syncthreads();

    // ---- consumer: each wave -> its 16 rows, 8 d-tiles ----
    v16bf A = *(const v16bf*)&Wop[((wave << 5) + lane) * 16];
#pragma unroll
    for (int t = 0; t < 8; ++t) {
      v16bf Bv = *(const v16bf*)&Yop[((t << 5) + lane) * 16];
      acc[t] = __builtin_amdgcn_wmma_f32_16x16x32_bf16(
          false, A, false, Bv, (short)0, acc[t], false, false);
    }
  }

  // ---- epilogue: sigmoid, C/D layout: VGPR r -> M = r + 8*(lane>=16) ----
  const int n  = lane & 15;
  const int mb = (lane >> 4) << 3;
#pragma unroll
  for (int t = 0; t < 8; ++t) {
#pragma unroll
    for (int r = 0; r < 8; ++r) {
      int i = I0 + (wave << 4) + mb + r;
      float v = acc[t][r];
      out[((size_t)b * NN + i) * DD + (t << 4) + n] = 1.f / (1.f + __expf(-v));
    }
  }
}

// ---------------------------------------------------------------------------
extern "C" void kernel_launch(void* const* d_in, const int* in_sizes, int n_in,
                              void* d_out, int out_size, void* d_ws, size_t ws_size,
                              hipStream_t stream) {
  const float* x     = (const float*)d_in[0];
  const float* a     = (const float*)d_in[1];
  const float* gamma = (const float*)d_in[2];
  const float* beta  = (const float*)d_in[3];
  float* out = (float*)d_out;

  float* ws = (float*)d_ws;
  float* P  = ws;                       // B*N
  float* Q  = ws + BATCH * NN;          // B*N
  float* CM = ws + 2 * BATCH * NN;      // B*N  column max
  float* CI = ws + 3 * BATCH * NN;      // B*N  1/column sum

  k_stats<<<BATCH, 256, 0, stream>>>(x, a, P, Q);
  k_cols<<<BATCH, 256, 0, stream>>>(gamma, beta, P, Q, CM, CI);
  k_gemm<<<dim3(NN / 128, BATCH), 256, 0, stream>>>(x, gamma, beta, P, Q, CM, CI, out);
}
